// PolyhedronResidualModel_62242666053928
// MI455X (gfx1250) — matrix operand order repaired and checked by reading, not compiled, not run on hardware
//
#include <hip/hip_runtime.h>

#define N_NODES  50000
#define N_EDGES  800000
#define D_NODE   128
#define D_EDGE   64
#define ZIN      320           // 2*D_NODE + D_EDGE
#define N_GRAPHS 512
#define HDIM     256
#define ZSTR     328           // padded LDS row stride (elements), 656B = 41*16B

typedef __bf16 bf16;
typedef __attribute__((ext_vector_type(16))) __bf16 v16bf;
typedef __attribute__((ext_vector_type(8)))  __bf16 v8bf;
typedef __attribute__((ext_vector_type(4)))  __bf16 v4bf;
typedef __attribute__((ext_vector_type(8)))  float  v8f;

union BFrag { v16bf v; v8bf h[2]; };

__device__ __forceinline__ v8f wmma_bf16(const BFrag& a, const BFrag& b, v8f c) {
    // D = A(16x32 bf16) x B(32x16 bf16) + C(16x16 f32)
    return __builtin_amdgcn_wmma_f32_16x16x32_bf16(false, a.v, false, b.v,
                                                   (short)0, c, false, false);
}

__device__ __forceinline__ v4bf pack4(float4 v) {
    v4bf o;
    o[0] = (bf16)v.x; o[1] = (bf16)v.y; o[2] = (bf16)v.z; o[3] = (bf16)v.w;
    return o;
}

// ---------------------------------------------------------------- utilities
__global__ void cvt_f32_bf16_kernel(const float* __restrict__ s,
                                    bf16* __restrict__ d, int n) {
    int i = blockIdx.x * blockDim.x + threadIdx.x;
    if (i < n) d[i] = (bf16)s[i];
}

__global__ void zero_f32_kernel(float* __restrict__ p, int n) {
    int i = blockIdx.x * blockDim.x + threadIdx.x;
    if (i < n) p[i] = 0.0f;
}

// ---------------------------------------------------------- CGConv edge GEMM
// One 16-edge tile per loop iteration. Block = 256 threads = 8 waves;
// wave w computes output channels [16w, 16w+16). z tile (16x320 bf16) staged
// in LDS via branch-free vectorized gather. msg = sigmoid(z@Wf.T + bf) *
// softplus(z@Ws.T + bs), scatter-added into node_out (residual-initialized).
__global__ void __launch_bounds__(256)
cgconv_edge_kernel(const float* __restrict__ node_in,
                   const int*   __restrict__ srcI,
                   const int*   __restrict__ dstI,
                   const float* __restrict__ eattr,
                   const bf16*  __restrict__ Wf,   // [128][320] bf16 row-major
                   const bf16*  __restrict__ Ws,   // [128][320] bf16 row-major
                   const float* __restrict__ biasF,
                   const float* __restrict__ biasS,
                   float*       __restrict__ node_out,
                   int n_tiles) {
    __shared__ __align__(16) bf16 z[16 * ZSTR];
    __shared__ int dsts[16];
    __shared__ int srcs[16];

    const int t    = threadIdx.x;
    const int wave = t >> 5;          // 0..7  -> N-tile
    const int lane = t & 31;
    const int half = lane >> 4;       // 0/1
    const int l15  = lane & 15;
    const int ch   = wave * 16 + l15; // output channel owned by this lane
    const float bF = biasF[ch];
    const float bS = biasS[ch];

    // gather mapping: 16 threads per edge-row, 5 float4 loads each (branch-free)
    const int r = t >> 4;             // edge row in tile: 0..15
    const int q = t & 15;             // float4 slot within segment

    // B-fragment base: lane holds W[ch][kb*32 + half*8 .. +7] and W[ch][kb*32 + 16 + half*8 .. +7]
    const bf16* wf_base = Wf + (size_t)ch * ZIN + half * 8;
    const bf16* ws_base = Ws + (size_t)ch * ZIN + half * 8;
    // A-fragment base inside LDS z tile
    const bf16* za = &z[l15 * ZSTR + half * 8];

    for (int tile = blockIdx.x; tile < n_tiles; tile += gridDim.x) {
        const int e0 = tile * 16;
        if (t < 16)                dsts[t]      = dstI[e0 + t];
        else if (t < 32)           srcs[t - 16] = srcI[e0 + t - 16];
        __syncthreads();

        // cooperative gather: z[r] = [x[dst[r]] | x[src[r]] | eattr[r]] as bf16
        {
            const float4* pd = (const float4*)(node_in + (size_t)dsts[r] * D_NODE);
            const float4* ps = (const float4*)(node_in + (size_t)srcs[r] * D_NODE);
            const float4* pe = (const float4*)(eattr   + (size_t)(e0 + r) * D_EDGE);
            bf16* zr = &z[r * ZSTR];
            *(v4bf*)(zr + (q     ) * 4) = pack4(pd[q]);
            *(v4bf*)(zr + (q + 16) * 4) = pack4(pd[q + 16]);
            *(v4bf*)(zr + (q + 32) * 4) = pack4(ps[q]);
            *(v4bf*)(zr + (q + 48) * 4) = pack4(ps[q + 16]);
            *(v4bf*)(zr + (q + 64) * 4) = pack4(pe[q]);
        }
        __syncthreads();

        v8f accf = {};
        v8f accs = {};
        // double-buffered A fragments: overlap ds_load with WMMA
        BFrag aA, aB;
        aA.h[0] = *(const v8bf*)(za);
        aA.h[1] = *(const v8bf*)(za + 16);
#pragma unroll
        for (int kb = 0; kb < ZIN / 32; ++kb) {      // 10 K-blocks of 32
            BFrag& cur = (kb & 1) ? aB : aA;
            BFrag& nxt = (kb & 1) ? aA : aB;
            if (kb < ZIN / 32 - 1) {
                nxt.h[0] = *(const v8bf*)(za + (kb + 1) * 32);
                nxt.h[1] = *(const v8bf*)(za + (kb + 1) * 32 + 16);
            }
            BFrag bfm, bsm;   // loop-invariant across tiles -> hoisted to VGPRs
            bfm.h[0] = *(const v8bf*)(wf_base + kb * 32);
            bfm.h[1] = *(const v8bf*)(wf_base + kb * 32 + 16);
            bsm.h[0] = *(const v8bf*)(ws_base + kb * 32);
            bsm.h[1] = *(const v8bf*)(ws_base + kb * 32 + 16);
            accf = wmma_bf16(cur, bfm, accf);
            accs = wmma_bf16(cur, bsm, accs);
        }

        // gate + segment-sum scatter (by dst) with f32 atomics; branchless
#pragma unroll
        for (int v = 0; v < 8; ++v) {
            const float f = accf[v] + bF;
            const float s = accs[v] + bS;
            const float sig = 1.0f / (1.0f + __expf(-f));
            const float sp  = fmaxf(s, 0.0f) + log1pf(__expf(-fabsf(s)));
            const int m = half * 8 + v;               // edge row in tile
            atomicAdd(&node_out[(size_t)dsts[m] * D_NODE + ch], sig * sp);
        }
        __syncthreads();   // protect z/dsts/srcs reuse
    }
}

// ------------------------------------------------- per-node MLP + graph pool
// Block = 512 threads = 16 waves, one 16-node tile per block.
// y = x + Wp( relu(W1 relu(W0 x + b0) + b1) ) + bp ; pooled[batch[n]] += y
__global__ void __launch_bounds__(512)
node_mlp_pool_kernel(const float* __restrict__ node_in,
                     const int*   __restrict__ batch,
                     const bf16*  __restrict__ W0,   // [256][128]
                     const float* __restrict__ b0,
                     const bf16*  __restrict__ W1,   // [256][256]
                     const float* __restrict__ b1,
                     const bf16*  __restrict__ Wp,   // [128][256]
                     const float* __restrict__ bp,
                     float*       __restrict__ pooled) {
    __shared__ __align__(16) bf16 xb[16 * 136];   // x tile,  K=128, stride 136
    __shared__ __align__(16) bf16 h1[16 * 264];   // hidden1, K=256, stride 264
    __shared__ __align__(16) bf16 h2[16 * 264];   // hidden2
    __shared__ int bidx[16];

    const int t    = threadIdx.x;
    const int wave = t >> 5;          // 0..15
    const int lane = t & 31;
    const int half = lane >> 4;
    const int l15  = lane & 15;
    const int tb   = blockIdx.x * 16;

    // stage x tile (bf16, vectorized: 512 threads x 1 float4) + batch ids
    {
        const int m  = t >> 5;        // node row 0..15
        const int c4 = t & 31;        // float4 slot 0..31 (128 floats)
        const float4* px = (const float4*)(node_in + (size_t)(tb + m) * D_NODE);
        *(v4bf*)(&xb[m * 136 + c4 * 4]) = pack4(px[c4]);
    }
    if (t < 16) bidx[t] = batch[tb + t];
    __syncthreads();

    // ---- layer 0: h1 = relu(x @ W0.T + b0)   [16 x 256], 16 N-tiles
    {
        const int ch = wave * 16 + l15;
        const bf16* wb = W0 + (size_t)ch * 128 + half * 8;
        const bf16* xa = &xb[l15 * 136 + half * 8];
        v8f acc = {};
#pragma unroll
        for (int kb = 0; kb < 4; ++kb) {
            BFrag a, b;
            a.h[0] = *(const v8bf*)(xa + kb * 32);
            a.h[1] = *(const v8bf*)(xa + kb * 32 + 16);
            b.h[0] = *(const v8bf*)(wb + kb * 32);
            b.h[1] = *(const v8bf*)(wb + kb * 32 + 16);
            acc = wmma_bf16(a, b, acc);
        }
        const float bias = b0[ch];
#pragma unroll
        for (int v = 0; v < 8; ++v) {
            const int m = half * 8 + v;
            h1[m * 264 + ch] = (bf16)fmaxf(acc[v] + bias, 0.0f);
        }
    }
    __syncthreads();

    // ---- layer 1: h2 = relu(h1 @ W1.T + b1)  [16 x 256]
    {
        const int ch = wave * 16 + l15;
        const bf16* wb = W1 + (size_t)ch * 256 + half * 8;
        const bf16* xa = &h1[l15 * 264 + half * 8];
        v8f acc = {};
#pragma unroll
        for (int kb = 0; kb < 8; ++kb) {
            BFrag a, b;
            a.h[0] = *(const v8bf*)(xa + kb * 32);
            a.h[1] = *(const v8bf*)(xa + kb * 32 + 16);
            b.h[0] = *(const v8bf*)(wb + kb * 32);
            b.h[1] = *(const v8bf*)(wb + kb * 32 + 16);
            acc = wmma_bf16(a, b, acc);
        }
        const float bias = b1[ch];
#pragma unroll
        for (int v = 0; v < 8; ++v) {
            const int m = half * 8 + v;
            h2[m * 264 + ch] = (bf16)fmaxf(acc[v] + bias, 0.0f);
        }
    }
    __syncthreads();

    // ---- projection + residual + pool (waves 0..7 cover 128 channels)
    if (wave < 8) {
        const int ch = wave * 16 + l15;
        const bf16* wb = Wp + (size_t)ch * 256 + half * 8;
        const bf16* xa = &h2[l15 * 264 + half * 8];
        v8f acc = {};
#pragma unroll
        for (int kb = 0; kb < 8; ++kb) {
            BFrag a, b;
            a.h[0] = *(const v8bf*)(xa + kb * 32);
            a.h[1] = *(const v8bf*)(xa + kb * 32 + 16);
            b.h[0] = *(const v8bf*)(wb + kb * 32);
            b.h[1] = *(const v8bf*)(wb + kb * 32 + 16);
            acc = wmma_bf16(a, b, acc);
        }
        const float bias = bp[ch];
#pragma unroll
        for (int v = 0; v < 8; ++v) {
            const int m = half * 8 + v;
            const int n = tb + m;
            const float y = acc[v] + bias + node_in[(size_t)n * D_NODE + ch];
            atomicAdd(&pooled[(size_t)bidx[m] * D_NODE + ch], y);
        }
    }
}

// ------------------------------------------------------- graph-level head
// Negligible FLOPs (512 x ~131kFLOP): plain f32. One block per graph.
__global__ void __launch_bounds__(256)
head_kernel(const float* __restrict__ pooled,
            const float* __restrict__ W0, const float* __restrict__ b0,
            const float* __restrict__ W1, const float* __restrict__ b1,
            const float* __restrict__ Wp, const float* __restrict__ bp,
            const float* __restrict__ outW, const float* __restrict__ outb,
            float* __restrict__ out) {
    __shared__ float pr[128];
    __shared__ float h0[256];
    __shared__ float h1[256];
    __shared__ float yv[128];
    const int g = blockIdx.x;
    const int t = threadIdx.x;

    if (t < 128) pr[t] = pooled[(size_t)g * 128 + t];
    __syncthreads();

    { float a = b0[t];
      const float* w = W0 + (size_t)t * 128;
      for (int k = 0; k < 128; ++k) a += pr[k] * w[k];
      h0[t] = fmaxf(a, 0.0f); }
    __syncthreads();

    { float a = b1[t];
      const float* w = W1 + (size_t)t * 256;
      for (int k = 0; k < 256; ++k) a += h0[k] * w[k];
      h1[t] = fmaxf(a, 0.0f); }
    __syncthreads();

    if (t < 128) {
        float a = bp[t];
        const float* w = Wp + (size_t)t * 256;
        for (int k = 0; k < 256; ++k) a += h1[k] * w[k];
        a += pr[t];                    // residual
        yv[t] = a * outW[t];           // fold final linear's dot
    }
    __syncthreads();

    for (int s = 64; s > 0; s >>= 1) {
        if (t < s) yv[t] += yv[t + s];
        __syncthreads();
    }
    if (t == 0) out[g] = yv[0] + outb[0];
}

// --------------------------------------------------------------- launcher
extern "C" void kernel_launch(void* const* d_in, const int* in_sizes, int n_in,
                              void* d_out, int out_size, void* d_ws, size_t ws_size,
                              hipStream_t stream) {
    const float* x       = (const float*)d_in[0];
    const int*   eidx    = (const int*)  d_in[1];
    const float* eattr   = (const float*)d_in[2];
    const int*   batch   = (const int*)  d_in[3];
    const float* conv_Wf = (const float*)d_in[4];
    const float* conv_bf = (const float*)d_in[5];
    const float* conv_Ws = (const float*)d_in[6];
    const float* conv_bs = (const float*)d_in[7];
    const float* m1W0 = (const float*)d_in[8];
    const float* m1b0 = (const float*)d_in[9];
    const float* m1W1 = (const float*)d_in[10];
    const float* m1b1 = (const float*)d_in[11];
    const float* m1Wp = (const float*)d_in[12];
    const float* m1bp = (const float*)d_in[13];
    const float* m2W0 = (const float*)d_in[14];
    const float* m2b0 = (const float*)d_in[15];
    const float* m2W1 = (const float*)d_in[16];
    const float* m2b1 = (const float*)d_in[17];
    const float* m2Wp = (const float*)d_in[18];
    const float* m2bp = (const float*)d_in[19];
    const float* outW = (const float*)d_in[20];
    const float* outb = (const float*)d_in[21];
    float* out = (float*)d_out;

    char* ws = (char*)d_ws;
    size_t off = 0;
    auto take = [&](size_t bytes) {
        size_t o = off;
        off = (off + bytes + 255) & ~(size_t)255;
        return o;
    };
    bf16* wfB = (bf16*)(ws + take((size_t)2 * 128 * ZIN * sizeof(bf16)));
    bf16* wsB = (bf16*)(ws + take((size_t)2 * 128 * ZIN * sizeof(bf16)));
    bf16* w0B = (bf16*)(ws + take((size_t)256 * 128 * sizeof(bf16)));
    bf16* w1B = (bf16*)(ws + take((size_t)256 * 256 * sizeof(bf16)));
    bf16* wpB = (bf16*)(ws + take((size_t)128 * 256 * sizeof(bf16)));
    float* nb0    = (float*)(ws + take((size_t)N_NODES * D_NODE * sizeof(float)));
    float* nb1    = (float*)(ws + take((size_t)N_NODES * D_NODE * sizeof(float)));
    float* pooled = (float*)(ws + take((size_t)N_GRAPHS * D_NODE * sizeof(float)));

    // weight f32 -> bf16 conversions
    {
        int n = 2 * 128 * ZIN;
        cvt_f32_bf16_kernel<<<(n + 255) / 256, 256, 0, stream>>>(conv_Wf, wfB, n);
        cvt_f32_bf16_kernel<<<(n + 255) / 256, 256, 0, stream>>>(conv_Ws, wsB, n);
        n = 256 * 128;
        cvt_f32_bf16_kernel<<<(n + 255) / 256, 256, 0, stream>>>(m1W0, w0B, n);
        n = 256 * 256;
        cvt_f32_bf16_kernel<<<(n + 255) / 256, 256, 0, stream>>>(m1W1, w1B, n);
        n = 128 * 256;
        cvt_f32_bf16_kernel<<<(n + 255) / 256, 256, 0, stream>>>(m1Wp, wpB, n);
    }

    const int* srcI = eidx;             // edge_index[0]
    const int* dstI = eidx + N_EDGES;   // edge_index[1]
    const int n_etiles = N_EDGES / 16;  // 50000
    const size_t node_bytes = (size_t)N_NODES * D_NODE * sizeof(float);

    // layer 0: residual init then scatter-add messages
    hipMemcpyAsync(nb0, x, node_bytes, hipMemcpyDeviceToDevice, stream);
    cgconv_edge_kernel<<<4096, 256, 0, stream>>>(
        x, srcI, dstI, eattr, wfB, wsB, conv_bf, conv_bs, nb0, n_etiles);

    // layer 1
    hipMemcpyAsync(nb1, nb0, node_bytes, hipMemcpyDeviceToDevice, stream);
    cgconv_edge_kernel<<<4096, 256, 0, stream>>>(
        nb0, srcI, dstI, eattr, wfB + (size_t)128 * ZIN, wsB + (size_t)128 * ZIN,
        conv_bf + 128, conv_bs + 128, nb1, n_etiles);

    // node MLP + global_add_pool
    {
        int n = N_GRAPHS * D_NODE;
        zero_f32_kernel<<<(n + 255) / 256, 256, 0, stream>>>(pooled, n);
    }
    node_mlp_pool_kernel<<<N_NODES / 16, 512, 0, stream>>>(
        nb1, batch, w0B, m1b0, w1B, m1b1, wpB, m1bp, pooled);

    // graph head
    head_kernel<<<N_GRAPHS, 256, 0, stream>>>(
        pooled, m2W0, m2b0, m2W1, m2b1, m2Wp, m2bp, outW, outb, out);
}
